// CrossAttention_24395414242157
// MI455X (gfx1250) — compile-verified, matrix operand order checked by reading, not run on hardware
//
#include <hip/hip_runtime.h>
#include <hip/hip_bf16.h>

// Problem constants (from reference)
#define B_        16
#define N_        256
#define T_        1024
#define DRUG_DIM_ 768
#define TGT_DIM_  2560
#define H_        8
#define DH_       96
#define INNER_    768
#define EPS_      1e-5f

typedef __attribute__((ext_vector_type(16))) __bf16        v16bf;
typedef __attribute__((ext_vector_type(8)))  float         v8f;
typedef __attribute__((ext_vector_type(8)))  unsigned short v8u;

union BFV { v16bf bf; v8u u2[2]; };

__device__ __forceinline__ unsigned short f2bf(float f) {
    unsigned int u = __float_as_uint(f);
    unsigned int r = u + 0x7FFFu + ((u >> 16) & 1u);   // round-to-nearest-even
    return (unsigned short)(r >> 16);
}

// LDS byte offset (workgroup-relative) from a generic pointer to __shared__:
// generic = {shared_aperture_hi32, lds_offset_lo32}; DS/async ops take the low 32.
__device__ __forceinline__ unsigned lds_off(const void* p) {
    return (unsigned)(unsigned long long)p;
}

// async global -> LDS copy, 16B per lane (VGLOBAL GV mode, tracked by ASYNCcnt)
__device__ __forceinline__ void async_copy_b128(unsigned lds, const void* gptr) {
    asm volatile("global_load_async_to_lds_b128 %0, %1, off"
                 :: "v"(lds), "v"((unsigned long long)gptr) : "memory");
}

// ---------------------------------------------------------------------------
// fp32 -> bf16 elementwise convert (row-major)
// ---------------------------------------------------------------------------
__global__ void cvt_bf16_kernel(const float* __restrict__ src,
                                unsigned short* __restrict__ dst, long n) {
    long i = (long)blockIdx.x * blockDim.x + threadIdx.x;
    long stride = (long)gridDim.x * blockDim.x;
    for (; i < n; i += stride) dst[i] = f2bf(src[i]);
}

// fp32 [K x N] row-major -> bf16 [N x K] (transposed)
__global__ void cvt_bf16_T_kernel(const float* __restrict__ src,
                                  unsigned short* __restrict__ dst,
                                  int K, int Nn) {
    long total = (long)K * Nn;
    long i = (long)blockIdx.x * blockDim.x + threadIdx.x;
    long stride = (long)gridDim.x * blockDim.x;
    for (; i < total; i += stride) {
        int n = (int)(i / K);
        int k = (int)(i - (long)n * K);
        dst[i] = f2bf(src[(long)k * Nn + n]);
    }
}

// ===========================================================================
// LDS-tiled batched bf16 WMMA GEMM: C = alpha * A @ Bt^T
//   Workgroup tile 64(M) x 256(N), K-step 32, double-buffered async staging.
//   8 waves in 2(M) x 4(N) grid; each wave: 32x64 strip = 8 accumulators.
//   Branch-free hot loop: 5x global_load_async_to_lds_b128 (clamped prefetch
//   on the last step) -> s_wait_asynccnt 5 -> barrier -> 12x ds_load_b128 +
//   8x v_wmma_f32_16x16x32_bf16 -> barrier.
//   Requires M % 64 == 0, K % 32 == 0. N handled by clamp + store guards.
//   vmode==1: scatter-store for V projection -> vT[b][h][d][t]
// ===========================================================================
#define BM_   64
#define BN_   256
#define BK_   32
#define LSTR_ 40   // padded LDS row stride (elements); 80B -> conflict-free, 16B-aligned

__global__ __launch_bounds__(256) void gemm_bf16_wmma_tiled(
    const unsigned short* __restrict__ A,
    const unsigned short* __restrict__ Bt,
    float* __restrict__ Cf, unsigned short* __restrict__ Cb,
    int M, int Nd, int K, int lda, int ldb, int ldc,
    long aStrB, long aStrH, long bStrB, long bStrH, long cStrB, long cStrH,
    int Hn, float alpha, int vmode, int Tdim)
{
    __shared__ unsigned short Asm[2][BM_ * LSTR_];
    __shared__ unsigned short Bsm[2][BN_ * LSTR_];

    const int z  = blockIdx.y;
    const int bb = z / Hn;
    const int hh = z - bb * Hn;
    const unsigned short* Ab = A  + (long)bb * aStrB + (long)hh * aStrH;
    const unsigned short* Bb = Bt + (long)bb * bStrB + (long)hh * bStrH;
    const long cOff = (long)bb * cStrB + (long)hh * cStrH;

    const int nBlocks = (Nd + BN_ - 1) / BN_;
    const int bm = blockIdx.x / nBlocks;
    const int bn = blockIdx.x - bm * nBlocks;
    const int m0 = bm * BM_;
    const int n0 = bn * BN_;

    const int tid = threadIdx.x;

    // ---- async staging assignment: 256 threads, 16B each ----
    const int lrow = tid >> 2;            // 0..63
    const int lseg = tid & 3;             // 0..3 (8 elements each)
    const unsigned short* aG = Ab + (long)(m0 + lrow) * lda + lseg * 8;
    const unsigned short* bG[4];
#pragma unroll
    for (int r = 0; r < 4; ++r) {
        int c = n0 + lrow + r * 64;
        if (c > Nd - 1) c = Nd - 1;       // clamp: stores of these tiles are guarded
        bG[r] = Bb + (long)c * ldb + lseg * 8;
    }
    const unsigned aL = lds_off(&Asm[0][lrow * LSTR_ + lseg * 8]);
    unsigned bL[4];
#pragma unroll
    for (int r = 0; r < 4; ++r)
        bL[r] = lds_off(&Bsm[0][(lrow + r * 64) * LSTR_ + lseg * 8]);
    const unsigned aBuf = (unsigned)sizeof(Asm[0]);
    const unsigned bBuf = (unsigned)sizeof(Bsm[0]);

    // ---- compute-side fragment coordinates ----
    const int lane  = tid & 31;
    const int r15   = lane & 15;
    const int koff  = (lane >> 4) << 3;   // 16-bit A/B layout: runs at koff, koff+16
    const int hiOff = (lane >> 4) << 3;   // C/D layout: lanes 16-31 hold rows M+8
    const int wid   = tid >> 5;           // 0..7
    const int wm    = wid & 1;            // M half
    const int wn    = wid >> 1;           // 0..3
    const int mBase = wm * 32;
    const int nBase = wn * 64;

    v8f acc[2][4] = {{v8f{}, v8f{}, v8f{}, v8f{}}, {v8f{}, v8f{}, v8f{}, v8f{}}};

    const int nk = K / BK_;

    // prime buffer 0
    async_copy_b128(aL, aG);
    async_copy_b128(bL[0], bG[0]);
    async_copy_b128(bL[1], bG[1]);
    async_copy_b128(bL[2], bG[2]);
    async_copy_b128(bL[3], bG[3]);

    unsigned curA = 0, curB = 0;          // byte offsets of the active buffer
    for (int ki = 0; ki < nk; ++ki) {
        const unsigned nxtA = curA ^ aBuf;
        const unsigned nxtB = curB ^ bBuf;
        // unconditional prefetch; on the last step re-fetch the current k
        // into the dead buffer (never read; s_endpgm's wait-idle covers it)
        const long k1 = (ki + 1 < nk) ? (long)(ki + 1) * BK_ : (long)ki * BK_;
        async_copy_b128(aL + nxtA, aG + k1);
        async_copy_b128(bL[0] + nxtB, bG[0] + k1);
        async_copy_b128(bL[1] + nxtB, bG[1] + k1);
        async_copy_b128(bL[2] + nxtB, bG[2] + k1);
        async_copy_b128(bL[3] + nxtB, bG[3] + k1);
        // in-order completion: <=5 outstanding => current buffer's 5 are done
        asm volatile("s_wait_asynccnt 0x5" ::: "memory");
        __syncthreads();   // publish LDS writes to all waves

        const unsigned short* Ash = &Asm[0][0] + (curA >> 1);
        const unsigned short* Bsh = &Bsm[0][0] + (curB >> 1);

        BFV af[2];
#pragma unroll
        for (int mi = 0; mi < 2; ++mi) {
            const unsigned short* p = Ash + (mBase + mi * 16 + r15) * LSTR_ + koff;
            af[mi].u2[0] = *(const v8u*)(p);
            af[mi].u2[1] = *(const v8u*)(p + 16);
        }
#pragma unroll
        for (int j = 0; j < 4; ++j) {
            BFV bf_;
            const unsigned short* p = Bsh + (nBase + j * 16 + r15) * LSTR_ + koff;
            bf_.u2[0] = *(const v8u*)(p);
            bf_.u2[1] = *(const v8u*)(p + 16);
            acc[0][j] = __builtin_amdgcn_wmma_f32_16x16x32_bf16(
                false, af[0].bf, false, bf_.bf, (short)0, acc[0][j], false, false);
            acc[1][j] = __builtin_amdgcn_wmma_f32_16x16x32_bf16(
                false, af[1].bf, false, bf_.bf, (short)0, acc[1][j], false, false);
        }
        __syncthreads();   // all waves done reading before the buffer is reused
        curA = nxtA;
        curB = nxtB;
    }

    // ---- store 2x4 tiles of 16x16 ----
    if (vmode == 1) {
#pragma unroll
        for (int mi = 0; mi < 2; ++mi) {
#pragma unroll
            for (int j = 0; j < 4; ++j) {
                int cj = n0 + nBase + j * 16;
                if (cj + 16 > Nd) continue;
                int c  = cj + r15;
                int h2 = c / DH_;
                int dd = c - h2 * DH_;
#pragma unroll
                for (int r = 0; r < 8; ++r) {
                    int  m  = m0 + mBase + mi * 16 + r + hiOff;
                    int  bq = m / Tdim;
                    int  tt = m - bq * Tdim;
                    long ad = (((long)bq * H_ + h2) * DH_ + dd) * Tdim + tt;
                    Cb[ad] = f2bf(acc[mi][j][r] * alpha);
                }
            }
        }
    } else {
#pragma unroll
        for (int mi = 0; mi < 2; ++mi) {
#pragma unroll
            for (int j = 0; j < 4; ++j) {
                int cj = n0 + nBase + j * 16;
                if (cj + 16 > Nd) continue;
                long colBase = cOff + cj + r15;
#pragma unroll
                for (int r = 0; r < 8; ++r) {
                    int   m  = m0 + mBase + mi * 16 + r + hiOff;
                    long  ad = colBase + (long)m * ldc;
                    float v  = acc[mi][j][r] * alpha;
                    if (Cf) Cf[ad] = v;
                    if (Cb) Cb[ad] = f2bf(v);
                }
            }
        }
    }
}

// ===========================================================================
// Simple direct-load wave-tile GEMM (kept for Nd=96 attn@V where a 256-wide
// N block would waste ~60% of the compute). Wave = 16(M) x 64(N) strip.
// ===========================================================================
__global__ void gemm_bf16_wmma(
    const unsigned short* __restrict__ A,
    const unsigned short* __restrict__ Bt,
    float* __restrict__ Cf, unsigned short* __restrict__ Cb,
    int M, int Nd, int K, int lda, int ldb, int ldc,
    long aStrB, long aStrH, long bStrB, long bStrH, long cStrB, long cStrH,
    int Hn, float alpha)
{
    const int z  = blockIdx.y;
    const int bb = z / Hn;
    const int hh = z - bb * Hn;
    const unsigned short* Ab = A  + (long)bb * aStrB + (long)hh * aStrH;
    const unsigned short* Bb = Bt + (long)bb * bStrB + (long)hh * bStrH;
    const long cOff = (long)bb * cStrB + (long)hh * cStrH;

    const int nQuads = (Nd + 63) >> 6;
    const int total  = (M >> 4) * nQuads;
    const int wid    = blockIdx.x * (blockDim.x >> 5) + (threadIdx.x >> 5);
    if (wid >= total) return;

    const int lane  = threadIdx.x & 31;
    const int r15   = lane & 15;
    const int koff  = (lane >> 4) << 3;
    const int hiOff = (lane >> 4) << 3;

    const int mt = wid / nQuads;
    const int nq = wid - mt * nQuads;
    const int m0 = mt << 4;
    const int n0 = nq << 6;

    const unsigned short* aPtr = Ab + (long)(m0 + r15) * lda + koff;

    const unsigned short* bPtr[4];
    int cvalid[4], ccol[4];
#pragma unroll
    for (int j = 0; j < 4; ++j) {
        int cj    = n0 + (j << 4);
        cvalid[j] = (cj + 16 <= Nd);
        int cjb   = cvalid[j] ? cj : (Nd - 16);
        ccol[j]   = cjb;
        bPtr[j]   = Bb + (long)(cjb + r15) * ldb + koff;
    }

    v8f acc[4] = {v8f{}, v8f{}, v8f{}, v8f{}};

    for (int k0 = 0; k0 < K; k0 += 32) {
        BFV a;
        a.u2[0] = *(const v8u*)(aPtr);
        a.u2[1] = *(const v8u*)(aPtr + 16);
        aPtr += 32;

        BFV b0, b1, b2, b3;
        b0.u2[0] = *(const v8u*)(bPtr[0]); b0.u2[1] = *(const v8u*)(bPtr[0] + 16); bPtr[0] += 32;
        b1.u2[0] = *(const v8u*)(bPtr[1]); b1.u2[1] = *(const v8u*)(bPtr[1] + 16); bPtr[1] += 32;
        b2.u2[0] = *(const v8u*)(bPtr[2]); b2.u2[1] = *(const v8u*)(bPtr[2] + 16); bPtr[2] += 32;
        b3.u2[0] = *(const v8u*)(bPtr[3]); b3.u2[1] = *(const v8u*)(bPtr[3] + 16); bPtr[3] += 32;

        acc[0] = __builtin_amdgcn_wmma_f32_16x16x32_bf16(false, a.bf, false, b0.bf,
                                                         (short)0, acc[0], false, false);
        acc[1] = __builtin_amdgcn_wmma_f32_16x16x32_bf16(false, a.bf, false, b1.bf,
                                                         (short)0, acc[1], false, false);
        acc[2] = __builtin_amdgcn_wmma_f32_16x16x32_bf16(false, a.bf, false, b2.bf,
                                                         (short)0, acc[2], false, false);
        acc[3] = __builtin_amdgcn_wmma_f32_16x16x32_bf16(false, a.bf, false, b3.bf,
                                                         (short)0, acc[3], false, false);
    }

#pragma unroll
    for (int j = 0; j < 4; ++j) {
        if (!cvalid[j]) continue;
        long colBase = cOff + ccol[j] + r15;
#pragma unroll
        for (int r = 0; r < 8; ++r) {
            int   m  = m0 + r + hiOff;
            long  ad = colBase + (long)m * ldc;
            float v  = acc[j][r] * alpha;
            if (Cf) Cf[ad] = v;
            if (Cb) Cb[ad] = f2bf(v);
        }
    }
}

// ---------------------------------------------------------------------------
// Row softmax over T, then post-softmax masking (no renorm), dual-store
// fp32 (final attn output) + bf16 (operand for attn @ V).  1 block / row.
// ---------------------------------------------------------------------------
__global__ void softmax_mask_kernel(float* __restrict__ scores,
                                    unsigned short* __restrict__ attn_bf,
                                    const int* __restrict__ drug_mask,
                                    const int* __restrict__ pro_mask)
{
    const int z  = blockIdx.x;            // b*H*N + h*N + n
    const int n  = z % N_;
    const int bh = z / N_;
    const int b  = bh / H_;

    float* row = scores + (long)z * T_;
    unsigned short* orow = attn_bf + (long)z * T_;

    const int tid = threadIdx.x;          // 256 threads, 4 elems each
    float v[4];
    float m = -3.4e38f;
#pragma unroll
    for (int i = 0; i < 4; ++i) {
        v[i] = row[tid + i * 256];
        m = fmaxf(m, v[i]);
    }
#pragma unroll
    for (int off = 16; off > 0; off >>= 1) m = fmaxf(m, __shfl_xor(m, off, 32));

    __shared__ float red[8];
    const int w = tid >> 5, l = tid & 31;
    if (l == 0) red[w] = m;
    __syncthreads();
    float mm = red[0];
#pragma unroll
    for (int i = 1; i < 8; ++i) mm = fmaxf(mm, red[i]);
    __syncthreads();

    float s = 0.f;
#pragma unroll
    for (int i = 0; i < 4; ++i) { v[i] = __expf(v[i] - mm); s += v[i]; }
#pragma unroll
    for (int off = 16; off > 0; off >>= 1) s += __shfl_xor(s, off, 32);
    if (l == 0) red[w] = s;
    __syncthreads();
    float tot = 0.f;
#pragma unroll
    for (int i = 0; i < 8; ++i) tot += red[i];

    const float dm  = (drug_mask[b * N_ + n] != 0) ? 1.f : 0.f;
    const float inv = dm / tot;
#pragma unroll
    for (int i = 0; i < 4; ++i) {
        int   t  = tid + i * 256;
        float pm = (pro_mask[b * T_ + t] != 0) ? 1.f : 0.f;
        float o  = v[i] * inv * pm;
        row[t]  = o;
        orow[t] = f2bf(o);
    }
}

// ---------------------------------------------------------------------------
// x = proj + bo + drug ; LayerNorm(x)*gamma + beta.  1 block / row (768 cols).
// ---------------------------------------------------------------------------
__global__ void bias_res_ln_kernel(const float* __restrict__ proj,
                                   const float* __restrict__ drug,
                                   const float* __restrict__ bo,
                                   const float* __restrict__ gamma,
                                   const float* __restrict__ beta,
                                   float* __restrict__ out)
{
    const int  r   = blockIdx.x;
    const int  tid = threadIdx.x;         // 256 threads, 3 elems each
    const float* pr = proj + (long)r * DRUG_DIM_;
    const float* dr = drug + (long)r * DRUG_DIM_;
    float* o = out + (long)r * DRUG_DIM_;

    float x[3];
    float s = 0.f;
#pragma unroll
    for (int i = 0; i < 3; ++i) {
        int c = tid + i * 256;
        x[i] = pr[c] + bo[c] + dr[c];
        s += x[i];
    }
#pragma unroll
    for (int off = 16; off > 0; off >>= 1) s += __shfl_xor(s, off, 32);
    __shared__ float red[8];
    const int w = tid >> 5, l = tid & 31;
    if (l == 0) red[w] = s;
    __syncthreads();
    float tot = 0.f;
#pragma unroll
    for (int i = 0; i < 8; ++i) tot += red[i];
    const float mu = tot / (float)DRUG_DIM_;
    __syncthreads();

    float sq = 0.f;
#pragma unroll
    for (int i = 0; i < 3; ++i) { float d = x[i] - mu; sq += d * d; }
#pragma unroll
    for (int off = 16; off > 0; off >>= 1) sq += __shfl_xor(sq, off, 32);
    if (l == 0) red[w] = sq;
    __syncthreads();
    float vtot = 0.f;
#pragma unroll
    for (int i = 0; i < 8; ++i) vtot += red[i];
    const float rstd = rsqrtf(vtot / (float)DRUG_DIM_ + EPS_);

#pragma unroll
    for (int i = 0; i < 3; ++i) {
        int c = tid + i * 256;
        o[c] = (x[i] - mu) * rstd * gamma[c] + beta[c];
    }
}

// ---------------------------------------------------------------------------
// Host side
// ---------------------------------------------------------------------------
static inline void launch_gemm_tiled(hipStream_t s,
    const unsigned short* A, const unsigned short* Bt,
    float* Cf, unsigned short* Cb,
    int M, int Nd, int K, int lda, int ldb, int ldc,
    long aB, long aH, long bB, long bH, long cB, long cH,
    int Hn, int nz, float alpha, int vmode, int Tdim)
{
    int nB = (Nd + BN_ - 1) / BN_;
    dim3 grid((M / BM_) * nB, nz), block(256);
    gemm_bf16_wmma_tiled<<<grid, block, 0, s>>>(A, Bt, Cf, Cb, M, Nd, K, lda, ldb, ldc,
                                                aB, aH, bB, bH, cB, cH, Hn, alpha, vmode, Tdim);
}

static inline void launch_gemm_simple(hipStream_t s,
    const unsigned short* A, const unsigned short* Bt,
    float* Cf, unsigned short* Cb,
    int M, int Nd, int K, int lda, int ldb, int ldc,
    long aB, long aH, long bB, long bH, long cB, long cH,
    int Hn, int nz, float alpha)
{
    int nQuads = (Nd + 63) / 64;
    int total  = (M / 16) * nQuads;
    dim3 grid((total + 7) / 8, nz), block(256);
    gemm_bf16_wmma<<<grid, block, 0, s>>>(A, Bt, Cf, Cb, M, Nd, K, lda, ldb, ldc,
                                          aB, aH, bB, bH, cB, cH, Hn, alpha);
}

extern "C" void kernel_launch(void* const* d_in, const int* in_sizes, int n_in,
                              void* d_out, int out_size, void* d_ws, size_t ws_size,
                              hipStream_t stream)
{
    const float* drug      = (const float*)d_in[0];
    const float* target    = (const float*)d_in[1];
    const int*   drug_mask = (const int*)  d_in[2];
    const int*   pro_mask  = (const int*)  d_in[3];
    const float* Wq        = (const float*)d_in[4];
    const float* Wk        = (const float*)d_in[5];
    const float* Wv        = (const float*)d_in[6];
    const float* Wo        = (const float*)d_in[7];
    const float* bo        = (const float*)d_in[8];
    const float* gamma     = (const float*)d_in[9];
    const float* beta      = (const float*)d_in[10];

    float* out    = (float*)d_out;                           // (B,N,768)
    float* scores = out + (long)B_ * N_ * DRUG_DIM_;         // (B,H,N,T) attn output

    char* w = (char*)d_ws;
    auto carve = [&](size_t bytes) {
        void* p = (void*)w;
        w += (bytes + 255) & ~(size_t)255;
        return p;
    };
    unsigned short* drug_bf   = (unsigned short*)carve((size_t)B_ * N_ * DRUG_DIM_ * 2);
    unsigned short* target_bf = (unsigned short*)carve((size_t)B_ * T_ * TGT_DIM_ * 2);
    unsigned short* WqT = (unsigned short*)carve((size_t)INNER_ * DRUG_DIM_ * 2);
    unsigned short* WkT = (unsigned short*)carve((size_t)INNER_ * TGT_DIM_ * 2);
    unsigned short* WvT = (unsigned short*)carve((size_t)INNER_ * TGT_DIM_ * 2);
    unsigned short* WoT = (unsigned short*)carve((size_t)DRUG_DIM_ * INNER_ * 2);
    unsigned short* q_bf  = (unsigned short*)carve((size_t)B_ * N_ * INNER_ * 2);
    unsigned short* k_bf  = (unsigned short*)carve((size_t)B_ * T_ * INNER_ * 2);
    unsigned short* vT_bf = (unsigned short*)carve((size_t)B_ * H_ * DH_ * T_ * 2);
    unsigned short* attn_bf = (unsigned short*)carve((size_t)B_ * H_ * N_ * T_ * 2);
    unsigned short* ctx_bf  = (unsigned short*)carve((size_t)B_ * N_ * INNER_ * 2);
    float*          proj    = (float*)carve((size_t)B_ * N_ * DRUG_DIM_ * 4);

    // 1) fp32 -> bf16 conversions
    cvt_bf16_kernel<<<dim3(2048), dim3(256), 0, stream>>>(drug, drug_bf,
                                                          (long)B_ * N_ * DRUG_DIM_);
    cvt_bf16_kernel<<<dim3(8192), dim3(256), 0, stream>>>(target, target_bf,
                                                          (long)B_ * T_ * TGT_DIM_);
    cvt_bf16_T_kernel<<<dim3(2048), dim3(256), 0, stream>>>(Wq, WqT, DRUG_DIM_, INNER_);
    cvt_bf16_T_kernel<<<dim3(4096), dim3(256), 0, stream>>>(Wk, WkT, TGT_DIM_, INNER_);
    cvt_bf16_T_kernel<<<dim3(4096), dim3(256), 0, stream>>>(Wv, WvT, TGT_DIM_, INNER_);
    cvt_bf16_T_kernel<<<dim3(2048), dim3(256), 0, stream>>>(Wo, WoT, INNER_, DRUG_DIM_);

    // 2) Q = drug @ Wq  -> bf16 (B*N, INNER)
    launch_gemm_tiled(stream, drug_bf, WqT, nullptr, q_bf,
                      B_ * N_, INNER_, DRUG_DIM_, DRUG_DIM_, DRUG_DIM_, INNER_,
                      0, 0, 0, 0, 0, 0, 1, 1, 1.0f, 0, 1);

    // 3) K = target @ Wk -> bf16 (B*T, INNER)
    launch_gemm_tiled(stream, target_bf, WkT, nullptr, k_bf,
                      B_ * T_, INNER_, TGT_DIM_, TGT_DIM_, TGT_DIM_, INNER_,
                      0, 0, 0, 0, 0, 0, 1, 1, 1.0f, 0, 1);

    // 4) V = target @ Wv -> bf16 scattered to vT[b][h][d][t]
    launch_gemm_tiled(stream, target_bf, WvT, nullptr, vT_bf,
                      B_ * T_, INNER_, TGT_DIM_, TGT_DIM_, TGT_DIM_, 0,
                      0, 0, 0, 0, 0, 0, 1, 1, 1.0f, 1, T_);

    // 5) scores[b,h] = scale * q[b,:,h,:] @ k[b,:,h,:]^T  -> fp32 into d_out
    const float scale = 1.0f / sqrtf((float)DH_);
    launch_gemm_tiled(stream, q_bf, k_bf, scores, nullptr,
                      N_, T_, DH_, INNER_, INNER_, T_,
                      (long)N_ * INNER_, (long)DH_,
                      (long)T_ * INNER_, (long)DH_,
                      (long)H_ * N_ * T_, (long)N_ * T_,
                      H_, B_ * H_, scale, 0, 1);

    // 6) softmax + post-softmax masking; fp32 in place + bf16 for next GEMM
    softmax_mask_kernel<<<dim3(B_ * H_ * N_), dim3(256), 0, stream>>>(
        scores, attn_bf, drug_mask, pro_mask);

    // 7) ctx[b,n,h*DH+d] = attn[b,h] @ vT[b,h]^T  -> bf16 (Nd=96: simple kernel)
    launch_gemm_simple(stream, attn_bf, vT_bf, nullptr, ctx_bf,
                       N_, DH_, T_, T_, T_, INNER_,
                       (long)H_ * N_ * T_, (long)N_ * T_,
                       (long)H_ * DH_ * T_, (long)DH_ * T_,
                       (long)N_ * INNER_, (long)DH_,
                       H_, B_ * H_, 1.0f);

    // 8) proj = ctx @ Wo -> fp32
    launch_gemm_tiled(stream, ctx_bf, WoT, proj, nullptr,
                      B_ * N_, DRUG_DIM_, INNER_, INNER_, INNER_, DRUG_DIM_,
                      0, 0, 0, 0, 0, 0, 1, 1, 1.0f, 0, 1);

    // 9) bias + residual + LayerNorm -> d_out
    bias_res_ln_kernel<<<dim3(B_ * N_), dim3(256), 0, stream>>>(
        proj, drug, bo, gamma, beta, out);
}
